// TransMIL_46600395161722
// MI455X (gfx1250) — matrix operand order repaired and checked by reading, not compiled, not run on hardware
//
#include <hip/hip_runtime.h>
#include <hip/hip_bf16.h>

// ---------------------------------------------------------------------------
// TransMIL forward for MI455X (gfx1250).  All large GEMMs run through a
// batched WMMA f16->f32 kernel (v_wmma_f32_16x16x32_f16); everything else is
// lightweight VALU work.  Buffers live in d_ws; no allocations on stream.
// ---------------------------------------------------------------------------

typedef _Float16 v8h  __attribute__((ext_vector_type(8)));
typedef _Float16 v16h __attribute__((ext_vector_type(16)));
typedef float    v8f  __attribute__((ext_vector_type(8)));

#define NIN    16384      // tokens from input
#define INDIM  1024
#define CDIM   512
#define NTOK   16385      // + cls token
#define NPD    16640      // padded seq inside attention (pad=255 at front)
#define PADR   255
#define HEADS  8
#define DHD    64
#define MLM    256        // landmarks
#define LCH    65         // chunk length per landmark
#define HSP    128        // spatial H=W for PPEG

// ===========================================================================
// Batched GEMM:  C[M x N] (f32) = A[M x K] (f16, row-major, lda)
//                              @ B[K x N] (f16, row-major, ldb)
// batch = blockIdx.z with element strides sA/sB/sC.
// Block tile 128x64, K-step 32; 8 waves, each wave owns a 32x32 output tile
// (2x2 WMMA 16x16x32 -> 4 matrix ops per K-step, 2x operand reuse).
// Requires K % 32 == 0 (true for every call here).
// ===========================================================================
#define LDSP 40   // padded LDS row stride (halfs)

__global__ __launch_bounds__(256) void gemm_f16_wmma(
    const _Float16* __restrict__ Ag, const _Float16* __restrict__ Bg,
    float* __restrict__ Cg, int Md, int Nd, int Kd,
    int lda, int ldb, int ldc,
    long long sA, long long sB, long long sC)
{
  __shared__ _Float16 lA[128 * LDSP];   // 10240 B
  __shared__ _Float16 lB[64 * LDSP];    //  5120 B

  const _Float16* A = Ag + (long long)blockIdx.z * sA;
  const _Float16* B = Bg + (long long)blockIdx.z * sB;
  float*          C = Cg + (long long)blockIdx.z * sC;

  const int bm = blockIdx.y * 128;
  const int bn = blockIdx.x * 64;
  const int tid  = threadIdx.x;
  const int wave = tid >> 5;
  const int lane = tid & 31;
  const int wm = (wave & 3) * 32;   // wave rows within block tile
  const int wn = (wave >> 2) * 32;  // wave cols within block tile

  v8f acc00 = {}, acc01 = {}, acc10 = {}, acc11 = {};

  // global->LDS staging assignments
  const int ar = tid >> 2;              // 0..63 (also +64)
  const int ac = (tid & 3) << 3;        // 0,8,16,24
  const int bk = tid & 31;              // k within K-tile
  const int bn8 = (tid >> 5) << 3;      // 0..56

  int gr0 = bm + ar;      if (gr0 > Md - 1) gr0 = Md - 1;
  int gr1 = bm + ar + 64; if (gr1 > Md - 1) gr1 = Md - 1;
  int gn  = bn + bn8;     if (gn  > Nd - 8) gn  = Nd - 8;

  for (int k0 = 0; k0 < Kd; k0 += 32) {
    // A tile 128x32 (row-major in LDS); two rows per thread
    *(v8h*)&lA[ar * LDSP + ac] =
        *(const v8h*)&A[(long long)gr0 * lda + k0 + ac];
    *(v8h*)&lA[(ar + 64) * LDSP + ac] =
        *(const v8h*)&A[(long long)gr1 * lda + k0 + ac];
    // B tile staged transposed: lB[n][k]
    v8h tb = *(const v8h*)&B[(long long)(k0 + bk) * ldb + gn];
#pragma unroll
    for (int j = 0; j < 8; ++j) lB[(bn8 + j) * LDSP + bk] = tb[j];

    // prefetch next K-tile into cache while WMMAs run (global_prefetch_b8)
    if (k0 + 32 < Kd) {
      __builtin_prefetch(&A[(long long)gr0 * lda + k0 + 32 + ac], 0, 3);
      __builtin_prefetch(&A[(long long)gr1 * lda + k0 + 32 + ac], 0, 3);
      __builtin_prefetch(&B[(long long)(k0 + 32 + bk) * ldb + gn], 0, 3);
    }
    __syncthreads();

    // A fragment per ISA 16-bit A 16x32 layout:
    //   lane<16: K 0..7 then 16..23 ; lane>=16: K 8..15 then 24..31
    const int lm = lane & 15;
    const int ab = (lane < 16) ? 0 : 8;
    const int kb = (lane < 16) ? 0 : 8;

    v16h av0, av1, bv0, bv1;
    {
      int m = wm + lm;
      v8h lo = *(const v8h*)&lA[m * LDSP + ab];
      v8h hi = *(const v8h*)&lA[m * LDSP + ab + 16];
      av0 = __builtin_shufflevector(lo, hi, 0,1,2,3,4,5,6,7,8,9,10,11,12,13,14,15);
      m = wm + 16 + lm;
      lo = *(const v8h*)&lA[m * LDSP + ab];
      hi = *(const v8h*)&lA[m * LDSP + ab + 16];
      av1 = __builtin_shufflevector(lo, hi, 0,1,2,3,4,5,6,7,8,9,10,11,12,13,14,15);
    }
    // B fragment: lane<16 -> K 0..7 then 16..23 of column n ; lane>=16 -> K 8..15 then 24..31
    {
      int n = wn + lm;
      v8h lo = *(const v8h*)&lB[n * LDSP + kb];
      v8h hi = *(const v8h*)&lB[n * LDSP + kb + 16];
      bv0 = __builtin_shufflevector(lo, hi, 0,1,2,3,4,5,6,7,8,9,10,11,12,13,14,15);
      n = wn + 16 + lm;
      lo = *(const v8h*)&lB[n * LDSP + kb];
      hi = *(const v8h*)&lB[n * LDSP + kb + 16];
      bv1 = __builtin_shufflevector(lo, hi, 0,1,2,3,4,5,6,7,8,9,10,11,12,13,14,15);
    }

    acc00 = __builtin_amdgcn_wmma_f32_16x16x32_f16(false, av0, false, bv0, (short)0, acc00, false, false);
    acc01 = __builtin_amdgcn_wmma_f32_16x16x32_f16(false, av0, false, bv1, (short)0, acc01, false, false);
    acc10 = __builtin_amdgcn_wmma_f32_16x16x32_f16(false, av1, false, bv0, (short)0, acc10, false, false);
    acc11 = __builtin_amdgcn_wmma_f32_16x16x32_f16(false, av1, false, bv1, (short)0, acc11, false, false);
    __syncthreads();
  }

  // C layout: VGPR r -> M = r (+8 for lanes>=16), N = lane&15
  const int colb = bn + wn + (lane & 15);
  const int rowb = bm + wm + ((lane < 16) ? 0 : 8);
#pragma unroll
  for (int r = 0; r < 8; ++r) {
    int row0 = rowb + r, row1 = rowb + 16 + r;
    if (row0 < Md) {
      if (colb < Nd)      C[(long long)row0 * ldc + colb]      = acc00[r];
      if (colb + 16 < Nd) C[(long long)row0 * ldc + colb + 16] = acc01[r];
    }
    if (row1 < Md) {
      if (colb < Nd)      C[(long long)row1 * ldc + colb]      = acc10[r];
      if (colb + 16 < Nd) C[(long long)row1 * ldc + colb + 16] = acc11[r];
    }
  }
}

// ===========================================================================
// Elementwise / reduction kernels
// ===========================================================================
__global__ void cast_f16(const float* __restrict__ in, _Float16* __restrict__ out,
                         long long n) {
  long long i = (long long)blockIdx.x * 256 + threadIdx.x;
  if (i < n) out[i] = (_Float16)in[i];
}

__global__ void build_h(const float* __restrict__ fc1o, const float* __restrict__ fc1b,
                        const float* __restrict__ cls, float* __restrict__ h) {
  long long idx = (long long)blockIdx.x * 256 + threadIdx.x;   // NTOK*512
  long long t = idx >> 9; int c = (int)(idx & 511);
  if (t == 0) { h[idx] = cls[c]; return; }
  float v = fc1o[(t - 1) * CDIM + c] + fc1b[c];
  h[idx] = v > 0.f ? v : 0.f;
}

__global__ __launch_bounds__(256) void ln_pad_cast(
    const float* __restrict__ h, const float* __restrict__ g,
    const float* __restrict__ b, _Float16* __restrict__ xp) {
  int row = blockIdx.x;                  // NPD rows
  int tid = threadIdx.x;
  _Float16* o = xp + (long long)row * CDIM;
  if (row < PADR) { o[tid] = (_Float16)0.f; o[tid + 256] = (_Float16)0.f; return; }
  const float* s = h + (long long)(row - PADR) * CDIM;
  float v0 = s[tid], v1 = s[tid + 256];
  __shared__ float red[256];
  red[tid] = v0 + v1; __syncthreads();
  for (int off = 128; off; off >>= 1) { if (tid < off) red[tid] += red[tid + off]; __syncthreads(); }
  float mu = red[0] * (1.f / 512.f); __syncthreads();
  float d0 = v0 - mu, d1 = v1 - mu;
  red[tid] = d0 * d0 + d1 * d1; __syncthreads();
  for (int off = 128; off; off >>= 1) { if (tid < off) red[tid] += red[tid + off]; __syncthreads(); }
  float rs = rsqrtf(red[0] * (1.f / 512.f) + 1e-5f);
  o[tid]       = (_Float16)(d0 * rs * g[tid] + b[tid]);
  o[tid + 256] = (_Float16)(d1 * rs * g[tid + 256] + b[tid + 256]);
}

__global__ void split_heads(const float* __restrict__ qkv, _Float16* __restrict__ qh,
                            _Float16* __restrict__ kT, _Float16* __restrict__ vh) {
  long long idx = (long long)blockIdx.x * 256 + threadIdx.x;   // 8*NPD*64
  int hh = (int)(idx / ((long long)NPD * DHD));
  long long rem = idx % ((long long)NPD * DHD);
  int i = (int)(rem >> 6), d = (int)(rem & 63);
  const float* row = qkv + (long long)i * 1536 + hh * 64 + d;
  qh[(long long)hh * NPD * DHD + (long long)i * DHD + d] = (_Float16)(row[0] * 0.125f);
  kT[(long long)hh * DHD * NPD + (long long)d * NPD + i] = (_Float16)row[512];
  vh[(long long)hh * NPD * DHD + (long long)i * DHD + d] = (_Float16)row[1024];
}

__global__ void landmarks_kernel(const float* __restrict__ qkv,
                                 _Float16* __restrict__ ql, _Float16* __restrict__ klT) {
  int idx = blockIdx.x * 256 + threadIdx.x;    // 8*256*64
  int hh = idx >> 14;
  int rem = idx & 16383;
  int i = rem >> 6, d = rem & 63;
  const float* base = qkv + (long long)(i * LCH) * 1536 + hh * 64 + d;
  float sq = 0.f, sk = 0.f;
  for (int j = 0; j < LCH; ++j) { sq += base[(long long)j * 1536]; sk += base[(long long)j * 1536 + 512]; }
  ql [hh * 16384 + i * 64 + d]  = (_Float16)(sq * (0.125f / (float)LCH));
  klT[hh * 16384 + d * 256 + i] = (_Float16)(sk * (1.f / (float)LCH));
}

__global__ __launch_bounds__(256) void softmax_rows(
    const float* __restrict__ in, _Float16* __restrict__ o16,
    float* __restrict__ o32, int cols) {
  long long row = blockIdx.x;
  const float* p = in + row * (long long)cols;
  int tid = threadIdx.x;
  __shared__ float red[256];
  float mx = -3.4e38f;
  for (int c = tid; c < cols; c += 256) mx = fmaxf(mx, p[c]);
  red[tid] = mx; __syncthreads();
  for (int off = 128; off; off >>= 1) { if (tid < off) red[tid] = fmaxf(red[tid], red[tid + off]); __syncthreads(); }
  float m = red[0]; __syncthreads();
  float s = 0.f;
  for (int c = tid; c < cols; c += 256) s += __expf(p[c] - m);
  red[tid] = s; __syncthreads();
  for (int off = 128; off; off >>= 1) { if (tid < off) red[tid] += red[tid + off]; __syncthreads(); }
  float inv = 1.f / red[0];
  for (int c = tid; c < cols; c += 256) {
    float v = __expf(p[c] - m) * inv;
    o16[row * (long long)cols + c] = (_Float16)v;
    if (o32) o32[row * (long long)cols + c] = v;
  }
}

// global (over all heads) max row-sum / col-sum for pinv init
__global__ __launch_bounds__(256) void pinv_prep(const float* __restrict__ a2,
                                                 float* __restrict__ sscal) {
  int tid = threadIdx.x;
  __shared__ float red[256];
  float rmax = 0.f, cmax = 0.f;
  for (int hh = 0; hh < HEADS; ++hh) {
    const float* a = a2 + (long long)hh * 65536;
    float rs = 0.f, cs = 0.f;
    for (int j = 0; j < 256; ++j) { rs += fabsf(a[tid * 256 + j]); cs += fabsf(a[j * 256 + tid]); }
    rmax = fmaxf(rmax, rs); cmax = fmaxf(cmax, cs);
  }
  red[tid] = rmax; __syncthreads();
  for (int off = 128; off; off >>= 1) { if (tid < off) red[tid] = fmaxf(red[tid], red[tid + off]); __syncthreads(); }
  float s1 = red[0]; __syncthreads();
  red[tid] = cmax; __syncthreads();
  for (int off = 128; off; off >>= 1) { if (tid < off) red[tid] = fmaxf(red[tid], red[tid + off]); __syncthreads(); }
  if (tid == 0) sscal[0] = 1.f / (red[0] * s1);
}

__global__ void z_init(const float* __restrict__ a2, const float* __restrict__ sscal,
                       float* __restrict__ z32, _Float16* __restrict__ z16) {
  long long idx = (long long)blockIdx.x * 256 + threadIdx.x;   // 8*65536
  int hh = (int)(idx >> 16);
  int r = (int)(idx & 65535);
  int i = r >> 8, j = r & 255;
  float v = a2[((long long)hh << 16) + j * 256 + i] * sscal[0];
  z32[idx] = v; z16[idx] = (_Float16)v;
}

// out = dc*I + s*in   (per 256x256 block), write f16 (+ optional f32)
__global__ void diag_axpy(const float* __restrict__ in, _Float16* __restrict__ o16,
                          float* __restrict__ o32, float dc, float s, long long total) {
  long long idx = (long long)blockIdx.x * 256 + threadIdx.x;
  if (idx >= total) return;
  int j = (int)(idx & 255), i = (int)((idx >> 8) & 255);
  float v = s * in[idx] + ((i == j) ? dc : 0.f);
  o16[idx] = (_Float16)v;
  if (o32) o32[idx] = v;
}

// depthwise 33-tap conv over sequence per head, add into outh
__global__ void res_add(float* __restrict__ outh, const float* __restrict__ qkv,
                        const float* __restrict__ res_w) {
  long long idx = (long long)blockIdx.x * 256 + threadIdx.x;   // 8*NPD*64
  int hh = (int)(idx / ((long long)NPD * DHD));
  long long rem = idx % ((long long)NPD * DHD);
  int i = (int)(rem >> 6), d = (int)(rem & 63);
  const float* rw = res_w + hh * 33;
  float acc = 0.f;
  for (int t = 0; t < 33; ++t) {
    int sidx = i + t - 16;
    if ((unsigned)sidx < (unsigned)NPD)
      acc += qkv[(long long)sidx * 1536 + 1024 + hh * 64 + d] * rw[t];
  }
  outh[idx] += acc;
}

__global__ void merge_heads(const float* __restrict__ outh, _Float16* __restrict__ om) {
  long long idx = (long long)blockIdx.x * 256 + threadIdx.x;   // NPD*512
  long long i = idx >> 9; int c = (int)(idx & 511);
  int hh = c >> 6, d = c & 63;
  om[idx] = (_Float16)outh[(long long)hh * NPD * DHD + i * DHD + d];
}

__global__ void resid_add(float* __restrict__ h, const float* __restrict__ proj,
                          const float* __restrict__ ob) {
  long long idx = (long long)blockIdx.x * 256 + threadIdx.x;   // NTOK*512
  long long t = idx >> 9; int c = (int)(idx & 511);
  h[idx] += proj[(long long)(PADR + t) * CDIM + c] + ob[c];
}

__global__ __launch_bounds__(256) void ppeg_kernel(
    const float* __restrict__ h1, float* __restrict__ h2,
    const float* __restrict__ w7, const float* __restrict__ b7,
    const float* __restrict__ w5, const float* __restrict__ b5,
    const float* __restrict__ w3, const float* __restrict__ b3) {
  long long idx = (long long)blockIdx.x * 256 + threadIdx.x;   // 512*128*128
  int c = (int)(idx >> 14);
  int rc = (int)(idx & 16383);
  int y = rc >> 7, x = rc & 127;
#define FEAT(r, cl) h1[((long long)(1 + (r) * HSP + (cl))) * CDIM + c]
  float acc = FEAT(y, x) + b7[c] + b5[c] + b3[c];
  const float* W7 = w7 + c * 49;
  for (int ky = 0; ky < 7; ++ky) { int iy = y + ky - 3; if ((unsigned)iy >= (unsigned)HSP) continue;
    for (int kx = 0; kx < 7; ++kx) { int ix = x + kx - 3; if ((unsigned)ix >= (unsigned)HSP) continue;
      acc += FEAT(iy, ix) * W7[ky * 7 + kx]; } }
  const float* W5 = w5 + c * 25;
  for (int ky = 0; ky < 5; ++ky) { int iy = y + ky - 2; if ((unsigned)iy >= (unsigned)HSP) continue;
    for (int kx = 0; kx < 5; ++kx) { int ix = x + kx - 2; if ((unsigned)ix >= (unsigned)HSP) continue;
      acc += FEAT(iy, ix) * W5[ky * 5 + kx]; } }
  const float* W3 = w3 + c * 9;
  for (int ky = 0; ky < 3; ++ky) { int iy = y + ky - 1; if ((unsigned)iy >= (unsigned)HSP) continue;
    for (int kx = 0; kx < 3; ++kx) { int ix = x + kx - 1; if ((unsigned)ix >= (unsigned)HSP) continue;
      acc += FEAT(iy, ix) * W3[ky * 3 + kx]; } }
#undef FEAT
  h2[((long long)(1 + y * HSP + x)) * CDIM + c] = acc;
}

__global__ void copy_cls(const float* __restrict__ h1, float* __restrict__ h2) {
  int i = blockIdx.x * 256 + threadIdx.x;
  if (i < CDIM) h2[i] = h1[i];
}

__global__ __launch_bounds__(256) void final_head(
    const float* __restrict__ h, const float* __restrict__ g,
    const float* __restrict__ b, const float* __restrict__ w,
    const float* __restrict__ fb, float* __restrict__ out) {
  int tid = threadIdx.x;
  __shared__ float red[256];
  float v0 = h[tid], v1 = h[tid + 256];
  red[tid] = v0 + v1; __syncthreads();
  for (int off = 128; off; off >>= 1) { if (tid < off) red[tid] += red[tid + off]; __syncthreads(); }
  float mu = red[0] * (1.f / 512.f); __syncthreads();
  float d0 = v0 - mu, d1 = v1 - mu;
  red[tid] = d0 * d0 + d1 * d1; __syncthreads();
  for (int off = 128; off; off >>= 1) { if (tid < off) red[tid] += red[tid + off]; __syncthreads(); }
  float rs = rsqrtf(red[0] * (1.f / 512.f) + 1e-5f); __syncthreads();
  float l0 = d0 * rs * g[tid] + b[tid];
  float l1 = d1 * rs * g[tid + 256] + b[tid + 256];
  for (int j = 0; j < 4; ++j) {
    red[tid] = l0 * w[tid * 4 + j] + l1 * w[(tid + 256) * 4 + j]; __syncthreads();
    for (int off = 128; off; off >>= 1) { if (tid < off) red[tid] += red[tid + off]; __syncthreads(); }
    if (tid == 0) out[j] = red[0] + fb[j];
    __syncthreads();
  }
}

// ===========================================================================
// Host orchestration
// ===========================================================================
static inline dim3 ggrid(int Md, int Nd, int batch) {
  return dim3((unsigned)((Nd + 63) / 64), (unsigned)((Md + 127) / 128), (unsigned)batch);
}
static inline unsigned nblk(long long total) { return (unsigned)((total + 255) / 256); }

struct Scratch {
  float *h1, *h2, *big;               // big: NPD*1536 f32 (qkv / fc1-out / proj)
  _Float16 *xh16, *w16;
  _Float16 *xp16, *qh16, *kT16, *vh16, *ql16, *klT16;
  float *a2_32; _Float16 *a2_16;
  float *z32; _Float16 *z16;
  float *az32; _Float16 *az16;
  float *t32;  _Float16 *m16;
  float *sscal;
  float *logits; _Float16 *probs16;
  float *w32; _Float16 *wh16;
  float *bw32; _Float16 *bw16;
  float *outh; _Float16 *outm16;
};

static void run_attention(float* hbuf, const float* ng, const float* nb,
                          const float* qkv_w, const float* out_w,
                          const float* out_b, const float* res_w,
                          const Scratch& S, hipStream_t st) {
  const long long P64  = (long long)NPD * DHD;       // per-head q/v stride
  const long long PT64 = (long long)DHD * NPD;       // per-head kT stride
  const long long PL   = (long long)NPD * MLM;       // per-head logits stride
  const long long TOT8 = 8LL * 65536;                // pinv elementwise count

  // LN + zero-pad + f16 cast
  ln_pad_cast<<<NPD, 256, 0, st>>>(hbuf, ng, nb, S.xp16);
  // qkv GEMM
  cast_f16<<<nblk(512LL * 1536), 256, 0, st>>>(qkv_w, S.w16, 512LL * 1536);
  gemm_f16_wmma<<<ggrid(NPD, 1536, 1), 256, 0, st>>>(
      S.xp16, S.w16, S.big, NPD, 1536, 512, 512, 1536, 1536, 0, 0, 0);
  // head split + landmarks
  split_heads<<<nblk(8LL * NPD * DHD), 256, 0, st>>>(S.big, S.qh16, S.kT16, S.vh16);
  landmarks_kernel<<<nblk(8LL * MLM * DHD), 256, 0, st>>>(S.big, S.ql16, S.klT16);
  // a2 = softmax(ql @ klT)
  gemm_f16_wmma<<<ggrid(MLM, MLM, HEADS), 256, 0, st>>>(
      S.ql16, S.klT16, S.t32, MLM, MLM, DHD, DHD, MLM, MLM,
      (long long)MLM * DHD, (long long)DHD * MLM, 65536LL);
  softmax_rows<<<8 * MLM, 256, 0, st>>>(S.t32, S.a2_16, S.a2_32, MLM);
  // pinv init (global scalars across heads)
  pinv_prep<<<1, 256, 0, st>>>(S.a2_32, S.sscal);
  z_init<<<nblk(TOT8), 256, 0, st>>>(S.a2_32, S.sscal, S.z32, S.z16);
  // 6 Newton iterations, batched over heads
  for (int it = 0; it < 6; ++it) {
    gemm_f16_wmma<<<ggrid(MLM, MLM, HEADS), 256, 0, st>>>(
        S.a2_16, S.z16, S.az32, MLM, MLM, MLM, MLM, MLM, MLM, 65536LL, 65536LL, 65536LL);
    diag_axpy<<<nblk(TOT8), 256, 0, st>>>(S.az32, S.az16, nullptr, 0.f, 1.f, TOT8);
    diag_axpy<<<nblk(TOT8), 256, 0, st>>>(S.az32, S.m16, nullptr, 7.f, -1.f, TOT8);
    gemm_f16_wmma<<<ggrid(MLM, MLM, HEADS), 256, 0, st>>>(
        S.az16, S.m16, S.t32, MLM, MLM, MLM, MLM, MLM, MLM, 65536LL, 65536LL, 65536LL);
    diag_axpy<<<nblk(TOT8), 256, 0, st>>>(S.t32, S.m16, nullptr, 15.f, -1.f, TOT8);
    gemm_f16_wmma<<<ggrid(MLM, MLM, HEADS), 256, 0, st>>>(
        S.az16, S.m16, S.t32, MLM, MLM, MLM, MLM, MLM, MLM, 65536LL, 65536LL, 65536LL);
    diag_axpy<<<nblk(TOT8), 256, 0, st>>>(S.t32, S.m16, nullptr, 13.f, -1.f, TOT8);
    gemm_f16_wmma<<<ggrid(MLM, MLM, HEADS), 256, 0, st>>>(
        S.z16, S.m16, S.t32, MLM, MLM, MLM, MLM, MLM, MLM, 65536LL, 65536LL, 65536LL);
    diag_axpy<<<nblk(TOT8), 256, 0, st>>>(S.t32, S.z16, S.z32, 0.f, 0.25f, TOT8);
  }
  // a3 = softmax(ql @ kT); w = a3 @ v
  gemm_f16_wmma<<<ggrid(MLM, NPD, HEADS), 256, 0, st>>>(
      S.ql16, S.kT16, S.logits, MLM, NPD, DHD, DHD, NPD, NPD,
      (long long)MLM * DHD, PT64, (long long)MLM * NPD);
  softmax_rows<<<8 * MLM, 256, 0, st>>>(S.logits, S.probs16, nullptr, NPD);
  gemm_f16_wmma<<<ggrid(MLM, DHD, HEADS), 256, 0, st>>>(
      S.probs16, S.vh16, S.w32, MLM, DHD, NPD, NPD, DHD, DHD,
      (long long)MLM * NPD, P64, (long long)MLM * DHD);
  cast_f16<<<nblk(8LL * MLM * DHD), 256, 0, st>>>(S.w32, S.wh16, 8LL * MLM * DHD);
  // bw = Z @ w   (associativity: out = a1 @ (Z @ (a3@v)))
  gemm_f16_wmma<<<ggrid(MLM, DHD, HEADS), 256, 0, st>>>(
      S.z16, S.wh16, S.bw32, MLM, DHD, MLM, MLM, DHD, DHD,
      65536LL, (long long)MLM * DHD, (long long)MLM * DHD);
  cast_f16<<<nblk(8LL * MLM * DHD), 256, 0, st>>>(S.bw32, S.bw16, 8LL * MLM * DHD);
  // a1 = softmax(q @ klT); outh = a1 @ bw
  gemm_f16_wmma<<<ggrid(NPD, MLM, HEADS), 256, 0, st>>>(
      S.qh16, S.klT16, S.logits, NPD, MLM, DHD, DHD, MLM, MLM,
      P64, (long long)DHD * MLM, PL);
  softmax_rows<<<8 * NPD, 256, 0, st>>>(S.logits, S.probs16, nullptr, MLM);
  gemm_f16_wmma<<<ggrid(NPD, DHD, HEADS), 256, 0, st>>>(
      S.probs16, S.bw16, S.outh, NPD, DHD, MLM, MLM, DHD, DHD,
      PL, (long long)MLM * DHD, P64);
  // depthwise seq-conv residual on v, merge heads, output projection
  res_add<<<nblk(8LL * NPD * DHD), 256, 0, st>>>(S.outh, S.big, res_w);
  merge_heads<<<nblk((long long)NPD * CDIM), 256, 0, st>>>(S.outh, S.outm16);
  cast_f16<<<nblk(512LL * 512), 256, 0, st>>>(out_w, S.w16, 512LL * 512);
  gemm_f16_wmma<<<ggrid(NPD, CDIM, 1), 256, 0, st>>>(
      S.outm16, S.w16, S.big, NPD, CDIM, CDIM, CDIM, CDIM, CDIM, 0, 0, 0);
  resid_add<<<nblk((long long)NTOK * CDIM), 256, 0, st>>>(hbuf, S.big, out_b);
}

extern "C" void kernel_launch(void* const* d_in, const int* in_sizes, int n_in,
                              void* d_out, int out_size, void* d_ws, size_t ws_size,
                              hipStream_t stream) {
  (void)in_sizes; (void)n_in; (void)out_size; (void)ws_size;
  const float* x       = (const float*)d_in[0];
  const float* fc1_w   = (const float*)d_in[1];
  const float* fc1_b   = (const float*)d_in[2];
  const float* cls     = (const float*)d_in[3];
  const float* l1_ng   = (const float*)d_in[4];
  const float* l1_nb   = (const float*)d_in[5];
  const float* l1_qkvw = (const float*)d_in[6];
  const float* l1_outw = (const float*)d_in[7];
  const float* l1_outb = (const float*)d_in[8];
  const float* l1_resw = (const float*)d_in[9];
  const float* l2_ng   = (const float*)d_in[10];
  const float* l2_nb   = (const float*)d_in[11];
  const float* l2_qkvw = (const float*)d_in[12];
  const float* l2_outw = (const float*)d_in[13];
  const float* l2_outb = (const float*)d_in[14];
  const float* l2_resw = (const float*)d_in[15];
  const float* w7 = (const float*)d_in[16]; const float* b7 = (const float*)d_in[17];
  const float* w5 = (const float*)d_in[18]; const float* b5 = (const float*)d_in[19];
  const float* w3 = (const float*)d_in[20]; const float* b3 = (const float*)d_in[21];
  const float* norm_g = (const float*)d_in[22];
  const float* norm_b = (const float*)d_in[23];
  const float* fc2_w  = (const float*)d_in[24];
  const float* fc2_b  = (const float*)d_in[25];

  // ---- carve workspace --------------------------------------------------
  char* p = (char*)d_ws;
  auto carve = [&](size_t bytes) -> void* {
    void* r = (void*)p; p += (bytes + 255) & ~(size_t)255; return r;
  };
  Scratch S;
  S.h1    = (float*)carve((size_t)NTOK * CDIM * 4);
  S.h2    = (float*)carve((size_t)NTOK * CDIM * 4);
  S.big   = (float*)carve((size_t)NPD * 1536 * 4);
  S.xh16  = (_Float16*)carve((size_t)NIN * INDIM * 2);
  S.w16   = (_Float16*)carve((size_t)512 * 1536 * 2);
  S.xp16  = (_Float16*)carve((size_t)NPD * CDIM * 2);
  S.qh16  = (_Float16*)carve((size_t)8 * NPD * DHD * 2);
  S.kT16  = (_Float16*)carve((size_t)8 * NPD * DHD * 2);
  S.vh16  = (_Float16*)carve((size_t)8 * NPD * DHD * 2);
  S.ql16  = (_Float16*)carve((size_t)8 * MLM * DHD * 2);
  S.klT16 = (_Float16*)carve((size_t)8 * MLM * DHD * 2);
  S.a2_32 = (float*)carve((size_t)8 * 65536 * 4);
  S.a2_16 = (_Float16*)carve((size_t)8 * 65536 * 2);
  S.z32   = (float*)carve((size_t)8 * 65536 * 4);
  S.z16   = (_Float16*)carve((size_t)8 * 65536 * 2);
  S.az32  = (float*)carve((size_t)8 * 65536 * 4);
  S.az16  = (_Float16*)carve((size_t)8 * 65536 * 2);
  S.t32   = (float*)carve((size_t)8 * 65536 * 4);
  S.m16   = (_Float16*)carve((size_t)8 * 65536 * 2);
  S.sscal = (float*)carve(256);
  S.logits  = (float*)carve((size_t)8 * NPD * MLM * 4);
  S.probs16 = (_Float16*)carve((size_t)8 * NPD * MLM * 2);
  S.w32   = (float*)carve((size_t)8 * MLM * DHD * 4);
  S.wh16  = (_Float16*)carve((size_t)8 * MLM * DHD * 2);
  S.bw32  = (float*)carve((size_t)8 * MLM * DHD * 4);
  S.bw16  = (_Float16*)carve((size_t)8 * MLM * DHD * 2);
  S.outh  = (float*)carve((size_t)8 * NPD * DHD * 4);
  S.outm16= (_Float16*)carve((size_t)NPD * CDIM * 2);

  // ---- fc1 + ReLU + cls concat -----------------------------------------
  cast_f16<<<nblk((long long)NIN * INDIM), 256, 0, stream>>>(x, S.xh16, (long long)NIN * INDIM);
  cast_f16<<<nblk((long long)INDIM * CDIM), 256, 0, stream>>>(fc1_w, S.w16, (long long)INDIM * CDIM);
  gemm_f16_wmma<<<ggrid(NIN, CDIM, 1), 256, 0, stream>>>(
      S.xh16, S.w16, S.big, NIN, CDIM, INDIM, INDIM, CDIM, CDIM, 0, 0, 0);
  build_h<<<nblk((long long)NTOK * CDIM), 256, 0, stream>>>(S.big, fc1_b, cls, S.h1);

  // ---- attention layer 1 (residual into h1) -----------------------------
  run_attention(S.h1, l1_ng, l1_nb, l1_qkvw, l1_outw, l1_outb, l1_resw, S, stream);

  // ---- PPEG (h1 -> h2) ---------------------------------------------------
  ppeg_kernel<<<nblk(512LL * HSP * HSP), 256, 0, stream>>>(S.h1, S.h2, w7, b7, w5, b5, w3, b3);
  copy_cls<<<2, 256, 0, stream>>>(S.h1, S.h2);

  // ---- attention layer 2 (residual into h2) -----------------------------
  run_attention(S.h2, l2_ng, l2_nb, l2_qkvw, l2_outw, l2_outb, l2_resw, S, stream);

  // ---- final LN(cls) + fc2 ----------------------------------------------
  final_head<<<1, 256, 0, stream>>>(S.h2, norm_g, norm_b, fc2_w, fc2_b, (float*)d_out);
}
